// DCell_28896539967762
// MI455X (gfx1250) — matrix operand order repaired and checked by reading, not compiled, not run on hardware
//
#include <hip/hip_runtime.h>
#include <cstddef>

typedef __attribute__((ext_vector_type(16))) _Float16 v16h;
typedef __attribute__((ext_vector_type(8)))  _Float16 v8h;
typedef __attribute__((ext_vector_type(8)))  float    v8f;

#define BATCH 256
#define DOUT  20
#define NGENE 64
#define NPAD  32   // padded N (DOUT 20 -> 32, two 16-wide WMMA tiles)

__device__ __forceinline__ v8f wmma16x16x32(v16h a, v16h b, v8f c) {
    // (neg_a, A, neg_b, B, c_mod, C, reuse_a, reuse_b)
    return __builtin_amdgcn_wmma_f32_16x16x32_f16(false, a, false, b, (short)0, c,
                                                  false, false);
}

// A fragment, 16x32 f16 (ISA 7.12.2): lane = l + 16*hi holds row M=l,
// halves 0..7 -> K = kb + hi*8 + j ; halves 8..15 -> K = kb + 16 + hi*8 + j.
template<int K_PAD>
__device__ __forceinline__ v16h load_a_frag(const _Float16* Xs, int row, int kb, int hi) {
    const _Float16* p0 = Xs + (size_t)row * K_PAD + kb + hi * 8;
    v8h c0 = *(const v8h*)p0;
    v8h c1 = *(const v8h*)(p0 + 16);
    v16h a;
#pragma unroll
    for (int i = 0; i < 8; ++i) { a[i] = c0[i]; a[8 + i] = c1[i]; }
    return a;
}

// ---------------------------------------------------------------------------
// One workgroup per term: X[256 x K] @ W[K x 20] -> z[256 x 20],
// BatchNorm over the batch axis (biased var, eps=1e-5), tanh, f16 out.
// K_PAD: 64 (leaf: genes only) or 160 (mid: 4 children * 20 + 64 genes = 144).
// ---------------------------------------------------------------------------
template<int K_PAD, bool HAS_CHILD>
__global__ __launch_bounds__(256) void dcell_subsystem(
    const float* __restrict__ genes,      // [B, T, 64]
    const _Float16* __restrict__ hchild,  // [B, Tchild, 20] or nullptr
    const float* __restrict__ W,          // [T, din, 20]
    const float* __restrict__ bias,       // [T, 20]
    const float* __restrict__ gamma,      // [T, 20]
    const float* __restrict__ beta,       // [T, 20]
    int T, int Tchild,
    _Float16* __restrict__ hout)          // [B, T, 20]
{
    constexpr int DIN = HAS_CHILD ? (4 * DOUT + NGENE) : NGENE;
    static_assert((size_t)BATCH * K_PAD * sizeof(_Float16) >=
                  (size_t)BATCH * NPAD * sizeof(float), "z overlay fits");

    const int t   = blockIdx.x;
    const int tid = threadIdx.x;

    // X staging buffer, later overlaid with the f32 z tile (barrier-separated).
    __shared__ __align__(32) unsigned char smem_xz[(size_t)BATCH * K_PAD * sizeof(_Float16)];
    _Float16 (*Xs)[K_PAD] = reinterpret_cast<_Float16(*)[K_PAD]>(smem_xz);
    float    (*zb)[NPAD]  = reinterpret_cast<float(*)[NPAD]>(smem_xz);
    __shared__ __align__(32) _Float16 Wt[NPAD][K_PAD];   // W transposed [n][k]
    __shared__ float red_s[DOUT][8], red_q[DOUT][8];
    __shared__ float bn_scale[DOUT], bn_shift[DOUT];

    // ---- stage X (f16, zero-padded) -------------------------------------
    for (int idx = tid; idx < BATCH * K_PAD; idx += 256) {
        const int row = idx / K_PAD;
        const int k   = idx - row * K_PAD;
        _Float16 v = (_Float16)0.0f;
        if (HAS_CHILD) {
            if (k < 4 * DOUT)
                v = hchild[((size_t)row * Tchild + 4 * (size_t)t) * DOUT + k];
            else if (k < DIN)
                v = (_Float16)genes[((size_t)row * T + t) * NGENE + (k - 4 * DOUT)];
        } else {
            if (k < DIN)
                v = (_Float16)genes[((size_t)row * T + t) * NGENE + k];
        }
        Xs[row][k] = v;
    }
    // ---- stage W^T (f16, zero-padded) -----------------------------------
    for (int idx = tid; idx < NPAD * K_PAD; idx += 256) {
        const int n = idx / K_PAD;
        const int k = idx - n * K_PAD;
        _Float16 v = (_Float16)0.0f;
        if (n < DOUT && k < DIN)
            v = (_Float16)W[((size_t)t * DIN + k) * DOUT + n];
        Wt[n][k] = v;
    }
    __syncthreads();

    // ---- WMMA: each wave owns 32 batch rows (two M-tiles) ---------------
    const int wv   = tid >> 5;
    const int lane = tid & 31;
    const int l    = lane & 15;
    const int hi   = lane >> 4;
    const int mbase = wv * 32;

    v8f acc00 = {}; v8f acc01 = {}; v8f acc10 = {}; v8f acc11 = {};
#pragma unroll
    for (int kb = 0; kb < K_PAD; kb += 32) {
        v16h a0 = load_a_frag<K_PAD>(&Xs[0][0], mbase + l,      kb, hi);
        v16h a1 = load_a_frag<K_PAD>(&Xs[0][0], mbase + 16 + l, kb, hi);
        // B fragment: lane l (+16*hi) holds column N=l(+nb), K same as A fragment (kb + hi*8 + j, kb + 16 + hi*8 + j)
        v16h b0 = load_a_frag<K_PAD>(&Wt[0][0], l,      kb, hi);
        v16h b1 = load_a_frag<K_PAD>(&Wt[0][0], 16 + l, kb, hi);
        acc00 = wmma16x16x32(a0, b0, acc00);
        acc01 = wmma16x16x32(a0, b1, acc01);
        acc10 = wmma16x16x32(a1, b0, acc10);
        acc11 = wmma16x16x32(a1, b1, acc11);
    }
    __syncthreads();   // all waves done reading Xs; safe to overlay z

    // ---- write z (+bias) to LDS; C layout: elem r -> (M = hi*8+r, N = l) -
    {
        const float* bv = bias + (size_t)t * DOUT;
        const int c0 = l, c1 = 16 + l;
        const float b0v = (c0 < DOUT) ? bv[c0] : 0.0f;
        const float b1v = (c1 < DOUT) ? bv[c1] : 0.0f;
#pragma unroll
        for (int r = 0; r < 8; ++r) {
            zb[mbase +      hi * 8 + r][c0] = acc00[r] + b0v;
            zb[mbase +      hi * 8 + r][c1] = acc01[r] + b1v;
            zb[mbase + 16 + hi * 8 + r][c0] = acc10[r] + b0v;
            zb[mbase + 16 + hi * 8 + r][c1] = acc11[r] + b1v;
        }
    }
    __syncthreads();

    // ---- BatchNorm stats over the 256-row batch -------------------------
    if (tid < DOUT * 8) {
        const int o = tid >> 3, c = tid & 7;
        float s = 0.f, q = 0.f;
#pragma unroll 4
        for (int r = 0; r < 32; ++r) {
            const float z = zb[c * 32 + r][o];
            s += z; q += z * z;
        }
        red_s[o][c] = s; red_q[o][c] = q;
    }
    __syncthreads();
    if (tid < DOUT) {
        float s = 0.f, q = 0.f;
#pragma unroll
        for (int c = 0; c < 8; ++c) { s += red_s[tid][c]; q += red_q[tid][c]; }
        const float mu  = s * (1.0f / BATCH);
        const float var = q * (1.0f / BATCH) - mu * mu;
        const float rs  = rsqrtf(var + 1e-5f);
        const float sc  = gamma[(size_t)t * DOUT + tid] * rs;
        bn_scale[tid] = sc;
        bn_shift[tid] = beta[(size_t)t * DOUT + tid] - mu * sc;
    }
    __syncthreads();

    // ---- tanh + f16 store: thread tid owns batch row tid ----------------
    {
        _Float16* ho = hout + ((size_t)tid * T + t) * DOUT;
#pragma unroll
        for (int o = 0; o < DOUT; ++o)
            ho[o] = (_Float16)tanhf(zb[tid][o] * bn_scale[o] + bn_shift[o]);
    }
}

// ---------------------------------------------------------------------------
// Root: x0 = [h1 (256x2560 f16) | genes0 (256x64)] @ W0[2624x20] -> BN -> tanh
// -> head dot(hw0[20]) + hb0.  One block, thread b owns batch row b.
// ---------------------------------------------------------------------------
__global__ __launch_bounds__(256) void dcell_root(
    const _Float16* __restrict__ h1,     // [B, 128, 20] = [B, 2560]
    const float* __restrict__ genes0,    // [B, 1, 64]
    const float* __restrict__ W0,        // [1, 2624, 20]
    const float* __restrict__ b0,        // [1, 20]
    const float* __restrict__ g0,        // [1, 20]
    const float* __restrict__ be0,       // [1, 20]
    const float* __restrict__ hw0,       // [1, 20, 1]
    const float* __restrict__ hb0,       // [1, 1]
    float* __restrict__ out)             // [B, 1]
{
    __shared__ float zb[BATCH][DOUT];
    __shared__ float red_s[DOUT][8], red_q[DOUT][8];
    __shared__ float bn_scale[DOUT], bn_shift[DOUT];
    const int tid = threadIdx.x;

    float acc[DOUT];
#pragma unroll
    for (int o = 0; o < DOUT; ++o) acc[o] = b0[o];

    const _Float16* x1 = h1 + (size_t)tid * 2560;
    for (int k = 0; k < 2560; ++k) {
        const float x = (float)x1[k];
        const float* wr = W0 + (size_t)k * DOUT;
#pragma unroll
        for (int o = 0; o < DOUT; ++o) acc[o] += x * wr[o];
    }
    const float* xg = genes0 + (size_t)tid * NGENE;
    for (int k = 0; k < NGENE; ++k) {
        const float x = xg[k];
        const float* wr = W0 + (size_t)(2560 + k) * DOUT;
#pragma unroll
        for (int o = 0; o < DOUT; ++o) acc[o] += x * wr[o];
    }
#pragma unroll
    for (int o = 0; o < DOUT; ++o) zb[tid][o] = acc[o];
    __syncthreads();

    if (tid < DOUT * 8) {
        const int o = tid >> 3, c = tid & 7;
        float s = 0.f, q = 0.f;
#pragma unroll 4
        for (int r = 0; r < 32; ++r) { const float z = zb[c * 32 + r][o]; s += z; q += z * z; }
        red_s[o][c] = s; red_q[o][c] = q;
    }
    __syncthreads();
    if (tid < DOUT) {
        float s = 0.f, q = 0.f;
#pragma unroll
        for (int c = 0; c < 8; ++c) { s += red_s[tid][c]; q += red_q[tid][c]; }
        const float mu  = s * (1.0f / BATCH);
        const float var = q * (1.0f / BATCH) - mu * mu;
        const float rs  = rsqrtf(var + 1e-5f);
        const float sc  = g0[tid] * rs;
        bn_scale[tid] = sc;
        bn_shift[tid] = be0[tid] - mu * sc;
    }
    __syncthreads();

    float p = hb0[0];
#pragma unroll
    for (int o = 0; o < DOUT; ++o)
        p += tanhf(zb[tid][o] * bn_scale[o] + bn_shift[o]) * hw0[o];
    out[tid] = p;
}

// ---------------------------------------------------------------------------
extern "C" void kernel_launch(void* const* d_in, const int* in_sizes, int n_in,
                              void* d_out, int out_size, void* d_ws, size_t ws_size,
                              hipStream_t stream) {
    (void)in_sizes; (void)n_in; (void)out_size; (void)ws_size;
    const float* genes3 = (const float*)d_in[0];
    const float* genes2 = (const float*)d_in[1];
    const float* genes1 = (const float*)d_in[2];
    const float* genes0 = (const float*)d_in[3];
    const float* W3 = (const float*)d_in[4];  const float* b3 = (const float*)d_in[5];
    const float* g3 = (const float*)d_in[6];  const float* be3 = (const float*)d_in[7];
    const float* W2 = (const float*)d_in[8];  const float* b2 = (const float*)d_in[9];
    const float* g2 = (const float*)d_in[10]; const float* be2 = (const float*)d_in[11];
    const float* W1 = (const float*)d_in[12]; const float* b1 = (const float*)d_in[13];
    const float* g1 = (const float*)d_in[14]; const float* be1 = (const float*)d_in[15];
    const float* W0 = (const float*)d_in[16]; const float* b0 = (const float*)d_in[17];
    const float* g0 = (const float*)d_in[18]; const float* be0 = (const float*)d_in[19];
    const float* hw0 = (const float*)d_in[20]; const float* hb0 = (const float*)d_in[21];

    // f16 intermediates in workspace: h3 (20.97 MB), h2 (5.24 MB), h1 (1.31 MB)
    _Float16* h3 = (_Float16*)d_ws;
    _Float16* h2 = h3 + (size_t)BATCH * 2048 * DOUT;
    _Float16* h1 = h2 + (size_t)BATCH * 512 * DOUT;

    dcell_subsystem<64, false><<<2048, 256, 0, stream>>>(
        genes3, nullptr, W3, b3, g3, be3, 2048, 0, h3);
    dcell_subsystem<160, true><<<512, 256, 0, stream>>>(
        genes2, h3, W2, b2, g2, be2, 512, 2048, h2);
    dcell_subsystem<160, true><<<128, 256, 0, stream>>>(
        genes1, h2, W1, b1, g1, be1, 128, 512, h1);
    dcell_root<<<1, 256, 0, stream>>>(
        h1, genes0, W0, b0, g0, be0, hw0, hb0, (float*)d_out);
}
